// GraphFeatureEncoder_69707319214253
// MI455X (gfx1250) — matrix operand order repaired and checked by reading, not compiled, not run on hardware
//
#include <hip/hip_runtime.h>
#include <hip/hip_bf16.h>
#include <math.h>

// ---------------------------------------------------------------------------
// GAT x3 + BN pipeline for MI455X (gfx1250, wave32, WMMA).
// GEMMs run on v_wmma_f32_16x16x32_f16 (f16 in, f32 accumulate). The weight
// matrix is pre-transposed once into f16 column-major, so every B fragment is
// two global_load_b128 of 8 contiguous K-halves each (hot in WGP$/L2: W is 128KB,
// reused by all 6250 blocks). A fragments are two global_load_b128 per K-step.
// Edge softmax/aggregation phase is memory-bound -> plain VMEM + atomics,
// which the 192MB L2 absorbs (whole 51MB feature matrix fits in L2).
// ---------------------------------------------------------------------------

typedef __attribute__((ext_vector_type(16))) _Float16 v16h;
typedef __attribute__((ext_vector_type(8)))  _Float16 v8h;
typedef __attribute__((ext_vector_type(8)))  float    v8f;

#define HEADS 4
#define LEAKY 0.2f
#define BN_EPS 1e-5f
#define SM_EPS 1e-16f

// ----- helpers --------------------------------------------------------------
__device__ __forceinline__ float lrelu(float x) {
    return x > 0.0f ? x : LEAKY * x;
}
// monotonic float <-> uint mapping for atomicMax-based segment max
__device__ __forceinline__ unsigned f2ord(float f) {
    unsigned u = __float_as_uint(f);
    return u ^ ((u >> 31) ? 0xFFFFFFFFu : 0x80000000u);
}
__device__ __forceinline__ float ord2f(unsigned u) {
    return __uint_as_float(u ^ ((u >> 31) ? 0x80000000u : 0xFFFFFFFFu));
}

// ----- f32 -> f16 conversion ------------------------------------------------
__global__ void cvt_f16_kernel(const float* __restrict__ in,
                               _Float16* __restrict__ out, int n) {
    int i = blockIdx.x * blockDim.x + threadIdx.x;
    if (i < n) out[i] = (_Float16)in[i];
}

// ----- f32 [K,Nc] -> f16 transposed [Nc,K] (weights, done once per layer) ---
__global__ void cvt_transpose_f16_kernel(const float* __restrict__ in,
                                         _Float16* __restrict__ out,
                                         int K, int Nc) {
    int t = blockIdx.x * blockDim.x + threadIdx.x;
    if (t >= K * Nc) return;
    int k = t / Nc, c = t - k * Nc;            // coalesced reads over c
    out[(size_t)c * K + k] = (_Float16)in[t];
}

// ----- WMMA GEMM: C[M,Nc] = A[M,K] @ B[K,Nc] with B given transposed -------
// grid: (M/16, Nc/128), block: 256 threads = 8 waves.
// Block tile 16(M) x 128(N); each wave owns one 16x16 N-subtile.
__global__ void gemm_wmma_kernel(const _Float16* __restrict__ A,   // [M,K]
                                 const _Float16* __restrict__ Bt,  // [Nc,K]
                                 float* __restrict__ C,
                                 int M, int K, int Nc) {
    const int lane  = threadIdx.x & 31;
    const int wave  = threadIdx.x >> 5;    // 0..7 -> N subtile
    const int hlane = lane & 15;
    const int hi16  = lane >> 4;           // 0 for lanes 0-15, 1 for 16-31

    const int row16 = blockIdx.x * 16;                         // M tile base
    const int col   = blockIdx.y * 128 + wave * 16 + hlane;    // output column
    const int arow  = row16 + hlane;
    const int aKoff = hi16 ? 8 : 0;        // A: lanes 16-31 carry K 8-15 / 24-31
    // B: lanes 0-15 carry K 0-7 / 16-23, lanes 16-31 carry K 8-15 / 24-31
    const _Float16* bcol = Bt + (size_t)col * K + (hi16 ? 8 : 0);
    const _Float16* apr  = A + (size_t)arow * K + aKoff;

    v8f acc = {};
    for (int k0 = 0; k0 < K; k0 += 32) {
        // A fragment: two contiguous 8-half runs per lane (16B aligned)
        v8h alo = *(const v8h*)(apr + k0);        // K = kbase + 0..7
        v8h ahi = *(const v8h*)(apr + k0 + 16);   // K = kbase + 16..23
        v16h a = __builtin_shufflevector(alo, ahi,
                 0,1,2,3,4,5,6,7,8,9,10,11,12,13,14,15);

        // B fragment: two 8-half runs (K +0..7, +16..23) from the transposed weight column
        v8h b0 = *(const v8h*)(bcol + k0);
        v8h b1 = *(const v8h*)(bcol + k0 + 16);
        v16h b = __builtin_shufflevector(b0, b1,
                 0,1,2,3,4,5,6,7,8,9,10,11,12,13,14,15);

        acc = __builtin_amdgcn_wmma_f32_16x16x32_f16(
                  false, a, false, b, (short)0, acc, false, false);
    }

    // D layout: VGPR v -> row (v or v+8), column = lane&15
    const int mbase = row16 + (hi16 ? 8 : 0);
    float* cp = C + (size_t)mbase * Nc + col;
#pragma unroll
    for (int v = 0; v < 8; ++v) cp[(size_t)v * Nc] = acc[v];
}

// ----- per-(node,head): attention dots + self-loop logit seeds max ---------
__global__ void attn_coef_kernel(const float* __restrict__ h,
                                 const float* __restrict__ a_src,
                                 const float* __restrict__ a_dst,
                                 float* __restrict__ als,
                                 float* __restrict__ ald,
                                 float* __restrict__ eself,
                                 unsigned* __restrict__ mbuf,
                                 int Nn, int C) {
    int t = blockIdx.x * blockDim.x + threadIdx.x;
    if (t >= Nn * HEADS) return;
    int node = t >> 2, head = t & 3;
    const float* hp = h + (size_t)node * (HEADS * C) + head * C;
    const float* as = a_src + head * C;
    const float* ad = a_dst + head * C;
    float ds = 0.0f, dd = 0.0f;
    for (int c = 0; c < C; ++c) { float v = hp[c]; ds += v * as[c]; dd += v * ad[c]; }
    als[t] = ds;
    ald[t] = dd;
    float es = lrelu(ds + dd);             // self-loop logit
    eself[t] = es;
    mbuf[t]  = f2ord(es);                  // seed segment max (every node has self-loop)
}

// ----- per-(edge,head): logit + atomic segment max -------------------------
__global__ void edge_max_kernel(const int* __restrict__ src,
                                const int* __restrict__ dst,
                                const float* __restrict__ als,
                                const float* __restrict__ ald,
                                float* __restrict__ ebuf,
                                unsigned* __restrict__ mbuf, int E) {
    int t = blockIdx.x * blockDim.x + threadIdx.x;
    if (t >= E * HEADS) return;
    int e = t >> 2, hh = t & 3;
    int s = src[e], d = dst[e];
    float v = lrelu(als[s * HEADS + hh] + ald[d * HEADS + hh]);
    ebuf[t] = v;
    atomicMax(&mbuf[d * HEADS + hh], f2ord(v));
}

// ----- per-(node,head): self-loop exp, init segment sum --------------------
__global__ void self_exp_kernel(const unsigned* __restrict__ mbuf,
                                float* __restrict__ eself,
                                float* __restrict__ sbuf, int Nn) {
    int t = blockIdx.x * blockDim.x + threadIdx.x;
    if (t >= Nn * HEADS) return;
    float p = __expf(eself[t] - ord2f(mbuf[t]));
    eself[t] = p;
    sbuf[t]  = p;
}

// ----- per-(edge,head): exp + atomic segment sum ---------------------------
__global__ void edge_exp_kernel(const int* __restrict__ dst,
                                float* __restrict__ ebuf,
                                const unsigned* __restrict__ mbuf,
                                float* __restrict__ sbuf, int E) {
    int t = blockIdx.x * blockDim.x + threadIdx.x;
    if (t >= E * HEADS) return;
    int e = t >> 2, hh = t & 3;
    int d = dst[e];
    float p = __expf(ebuf[t] - ord2f(mbuf[d * HEADS + hh]));
    ebuf[t] = p;
    atomicAdd(&sbuf[d * HEADS + hh], p);
}

// ----- per element: accumulator init with self-loop message ----------------
__global__ void agg_init_kernel(const float* __restrict__ h,
                                const float* __restrict__ eself,
                                const float* __restrict__ sbuf,
                                float* __restrict__ agg,
                                int Nn, int width, int C) {
    int t = blockIdx.x * blockDim.x + threadIdx.x;
    if (t >= Nn * width) return;
    int node = t / width, col = t - node * width, head = col / C;
    float a = eself[node * HEADS + head] / (sbuf[node * HEADS + head] + SM_EPS);
    agg[t] = a * h[t];
}

// ----- per-(edge,head): weighted scatter-add of source features ------------
__global__ void edge_agg_kernel(const int* __restrict__ src,
                                const int* __restrict__ dst,
                                const float* __restrict__ ebuf,
                                const float* __restrict__ sbuf,
                                const float* __restrict__ h,
                                float* __restrict__ agg,
                                int E, int width, int C) {
    int t = blockIdx.x * blockDim.x + threadIdx.x;
    if (t >= E * HEADS) return;
    int e = t >> 2, hh = t & 3;
    int s = src[e], d = dst[e];
    const float* hs = h   + (size_t)s * width + hh * C;
    float*       ad = agg + (size_t)d * width + hh * C;
    // prefetch the gathered source row (global_prefetch_b8) while we
    // compute the normalized coefficient
    __builtin_prefetch(hs, 0, 0);
    float a = ebuf[t] / (sbuf[d * HEADS + hh] + SM_EPS);
    for (int c = 0; c < C; ++c) atomicAdd(&ad[c], a * hs[c]);
}

// ----- bias + relu ----------------------------------------------------------
__global__ void bias_relu_kernel(const float* __restrict__ in,
                                 const float* __restrict__ b,
                                 float* __restrict__ out,
                                 int Nn, int width) {
    int t = blockIdx.x * blockDim.x + threadIdx.x;
    if (t >= Nn * width) return;
    int col = t % width;
    out[t] = fmaxf(in[t] + b[col], 0.0f);
}

// ----- batch norm (biased var) fused with f16 conversion for next GEMM -----
// one block per column
__global__ void bn_f16_kernel(const float* __restrict__ y,
                              const float* __restrict__ gamma,
                              const float* __restrict__ beta,
                              _Float16* __restrict__ xf16,
                              int Nn, int width) {
    __shared__ float ssum[256];
    __shared__ float ssq[256];
    int col = blockIdx.x;
    int tid = threadIdx.x;
    float s = 0.0f, q = 0.0f;
    for (int i = tid; i < Nn; i += blockDim.x) {
        float v = y[(size_t)i * width + col];
        s += v; q += v * v;
    }
    ssum[tid] = s; ssq[tid] = q;
    __syncthreads();
    for (int off = 128; off > 0; off >>= 1) {
        if (tid < off) { ssum[tid] += ssum[tid + off]; ssq[tid] += ssq[tid + off]; }
        __syncthreads();
    }
    float mean = ssum[0] / (float)Nn;
    float var  = ssq[0] / (float)Nn - mean * mean;
    float scale = rsqrtf(var + BN_EPS) * gamma[col];
    float shift = beta[col] - mean * scale;
    for (int i = tid; i < Nn; i += blockDim.x) {
        float v = y[(size_t)i * width + col];
        xf16[(size_t)i * width + col] = (_Float16)(v * scale + shift);
    }
}

// ---------------------------------------------------------------------------
extern "C" void kernel_launch(void* const* d_in, const int* in_sizes, int n_in,
                              void* d_out, int out_size, void* d_ws, size_t ws_size,
                              hipStream_t stream) {
    const float* x  = (const float*)d_in[0];
    const int*   ei = (const int*)d_in[1];

    const int Nn = in_sizes[0] / 128;   // 50000
    const int E  = in_sizes[1] / 2;     // 800000
    const int*   esrc = ei;
    const int*   edst = ei + E;

    // dict order: x, edge_index, (W,a_src,a_dst,b)x3, gamma1,beta1,gamma2,beta2
    const float* W[3]  = { (const float*)d_in[2],  (const float*)d_in[6],  (const float*)d_in[10] };
    const float* AS[3] = { (const float*)d_in[3],  (const float*)d_in[7],  (const float*)d_in[11] };
    const float* AD[3] = { (const float*)d_in[4],  (const float*)d_in[8],  (const float*)d_in[12] };
    const float* Bv[3] = { (const float*)d_in[5],  (const float*)d_in[9],  (const float*)d_in[13] };
    const float* GM[2] = { (const float*)d_in[14], (const float*)d_in[16] };
    const float* BT[2] = { (const float*)d_in[15], (const float*)d_in[17] };
    const int FIN[3] = { 128, 256, 256 };
    const int CH[3]  = { 64, 64, 32 };  // per-head out; width = 4*CH

    // ---- workspace carve-up -------------------------------------------------
    char* w = (char*)d_ws;
    auto carve = [&](size_t bytes) {
        void* p = (void*)w;
        w += (bytes + 255) & ~(size_t)255;
        return p;
    };
    _Float16* xf16  = (_Float16*)carve((size_t)Nn * 256 * sizeof(_Float16));
    _Float16* wf16t = (_Float16*)carve((size_t)256 * 256 * sizeof(_Float16));
    float*    hbuf  = (float*)carve((size_t)Nn * 256 * sizeof(float));
    float*    agg   = (float*)carve((size_t)Nn * 256 * sizeof(float));
    float*    als   = (float*)carve((size_t)Nn * HEADS * sizeof(float));
    float*    ald   = (float*)carve((size_t)Nn * HEADS * sizeof(float));
    float*    eself = (float*)carve((size_t)Nn * HEADS * sizeof(float));
    unsigned* mbuf  = (unsigned*)carve((size_t)Nn * HEADS * sizeof(unsigned));
    float*    sbuf  = (float*)carve((size_t)Nn * HEADS * sizeof(float));
    float*    ebuf  = (float*)carve((size_t)E * HEADS * sizeof(float));

    const int TB = 256;
    auto blocks = [&](long long n) { return (unsigned)((n + TB - 1) / TB); };

    // layer-1 input -> f16
    cvt_f16_kernel<<<blocks((long long)Nn * 128), TB, 0, stream>>>(x, xf16, Nn * 128);

    for (int l = 0; l < 3; ++l) {
        const int fin = FIN[l], C = CH[l], width = HEADS * C;

        // weights -> f16, transposed to [width, fin] (done once; ~65K elems)
        cvt_transpose_f16_kernel<<<blocks((long long)fin * width), TB, 0, stream>>>(
            W[l], wf16t, fin, width);

        // h = x @ W  (WMMA, B from pre-transposed weights)
        dim3 gg(Nn / 16, width / 128);
        gemm_wmma_kernel<<<gg, 256, 0, stream>>>(xf16, wf16t, hbuf, Nn, fin, width);

        // attention logits + softmax over incoming edges (incl. self-loop)
        attn_coef_kernel<<<blocks((long long)Nn * HEADS), TB, 0, stream>>>(
            hbuf, AS[l], AD[l], als, ald, eself, mbuf, Nn, C);
        edge_max_kernel<<<blocks((long long)E * HEADS), TB, 0, stream>>>(
            esrc, edst, als, ald, ebuf, mbuf, E);
        self_exp_kernel<<<blocks((long long)Nn * HEADS), TB, 0, stream>>>(
            mbuf, eself, sbuf, Nn);
        edge_exp_kernel<<<blocks((long long)E * HEADS), TB, 0, stream>>>(
            edst, ebuf, mbuf, sbuf, E);

        // message aggregation
        agg_init_kernel<<<blocks((long long)Nn * width), TB, 0, stream>>>(
            hbuf, eself, sbuf, agg, Nn, width, C);
        edge_agg_kernel<<<blocks((long long)E * HEADS), TB, 0, stream>>>(
            esrc, edst, ebuf, sbuf, hbuf, agg, E, width, C);

        // bias + relu
        float* target = (l == 2) ? (float*)d_out : agg;
        bias_relu_kernel<<<blocks((long long)Nn * width), TB, 0, stream>>>(
            agg, Bv[l], target, Nn, width);

        // batch norm + f16 conversion for next layer's GEMM input
        if (l < 2) {
            bn_f16_kernel<<<width, 256, 0, stream>>>(
                agg, GM[l], BT[l], xf16, Nn, width);
        }
    }
}
